// CPT_24111946400581
// MI455X (gfx1250) — matrix operand order repaired and checked by reading, not compiled, not run on hardware
//
#include <hip/hip_runtime.h>

#define B_ 8
#define C_ 128
#define N_ 20000
#define CHUNK 32
#define CPW 5            // chunks per workgroup: 125*5 = 625 = N/32 exactly
#define EPS_ 1e-5f

#define PCH_STRIDE 136   // halves; 272B rows: 16B aligned, bank-spread
#define QK_STRIDE  40    // halves; 80B rows: 16B aligned, bank-spread

typedef __attribute__((ext_vector_type(16))) _Float16 v16h;
typedef __attribute__((ext_vector_type(8)))  _Float16 v8h;
typedef __attribute__((ext_vector_type(8)))  float    v8f;
typedef __attribute__((ext_vector_type(4)))  float    v4f;

__device__ __forceinline__ v8f wmma16(v16h a, v16h b, v8f c) {
  return __builtin_amdgcn_wmma_f32_16x16x32_f16(false, a, false, b, (short)0, c, false, false);
}

__device__ __forceinline__ v16h cat16(v8h lo, v8h hi) {
  return __builtin_shufflevector(lo, hi, 0,1,2,3,4,5,6,7,8,9,10,11,12,13,14,15);
}
__device__ __forceinline__ v8h cvt8(v4f a, v4f b) {
  v8h r;
  r[0]=(_Float16)a[0]; r[1]=(_Float16)a[1]; r[2]=(_Float16)a[2]; r[3]=(_Float16)a[3];
  r[4]=(_Float16)b[0]; r[5]=(_Float16)b[1]; r[6]=(_Float16)b[2]; r[7]=(_Float16)b[3];
  return r;
}
// A-fragment from a global f32 row (16 contiguous floats, k = 16*lhalf + s)
__device__ __forceinline__ v16h afrag_global(const float* __restrict__ row, int kt, int lhalf) {
  const float* p0 = row + kt * 32 + lhalf * 16;
  const float* p1 = p0 + 8;
  v4f a0 = *(const v4f*)p0, a1 = *(const v4f*)(p0 + 4);
  v4f b0 = *(const v4f*)p1, b1 = *(const v4f*)(p1 + 4);
  return cat16(cvt8(a0, a1), cvt8(b0, b1));
}
// A-fragment from row-major LDS row
__device__ __forceinline__ v16h afrag_lds(const _Float16* row, int lhalf) {
  v8h lo = *(const v8h*)(row + lhalf * 16);
  v8h hi = *(const v8h*)(row + lhalf * 16 + 8);
  return cat16(lo, hi);
}
// B-fragment: 16 contiguous halves (K-major LDS tile)
__device__ __forceinline__ v16h bfrag_lds(const _Float16* p) {
  v8h lo = *(const v8h*)p;
  v8h hi = *(const v8h*)(p + 8);
  return cat16(lo, hi);
}
// pipelined pc loader: 16 coalesced b32 loads -> regs; later convert + 2x b128 LDS store
__device__ __forceinline__ void load_pc_regs(const float* __restrict__ src, float ld[16]) {
#pragma unroll
  for (int cc = 0; cc < 16; ++cc) ld[cc] = src[cc * N_];
}
__device__ __forceinline__ void store_pc_lds(_Float16* dst, const float ld[16]) {
  v8h p0, p1;
#pragma unroll
  for (int cc = 0; cc < 8; ++cc) p0[cc] = (_Float16)ld[cc];
#pragma unroll
  for (int cc = 0; cc < 8; ++cc) p1[cc] = (_Float16)ld[cc + 8];
  *(v8h*)dst       = p0;
  *(v8h*)(dst + 8) = p1;
}

// ---------------------------------------------------------------------------
// Kernel 1: fused q/k projections + BN + ReLU + graph-context + attn partial
// ---------------------------------------------------------------------------
__global__ __launch_bounds__(256)
void CPT_attn_accum(const float* __restrict__ pc, const float* __restrict__ x,
                    const float* __restrict__ Wq, const float* __restrict__ gq,
                    const float* __restrict__ bq, const float* __restrict__ mq,
                    const float* __restrict__ vq,
                    const float* __restrict__ Wk, const float* __restrict__ gk,
                    const float* __restrict__ bk, const float* __restrict__ mk,
                    const float* __restrict__ vk,
                    const float* __restrict__ Wg1, const float* __restrict__ bg1,
                    const float* __restrict__ Wg2, const float* __restrict__ bg2,
                    float* __restrict__ attn)
{
  __shared__ _Float16 pcht [2][CHUNK * PCH_STRIDE]; // pc chunk, K(=c)-major [n][c], ping-pong
  __shared__ _Float16 qlds [C_ * QK_STRIDE];        // qs chunk, row-major   [c][n]
  __shared__ _Float16 kglds[C_ * QK_STRIDE];        // k+gc chunk, row-major [d][n]

  const int b     = blockIdx.y;
  const int tid   = threadIdx.x;
  const int wave  = tid >> 5;
  const int lane  = tid & 31;
  const int lhalf = lane >> 4;
  const int l15   = lane & 15;

  const float* pcb = pc + (size_t)b * C_ * N_;
  const float* xb  = x  + (size_t)b * 4 * N_;
  const float* psrc = pcb + (size_t)(wave * 16) * N_ + (size_t)blockIdx.x * CPW * CHUNK + lane;

  // per-wave W A-fragments for o rows [16*wave, 16*wave+16)
  v16h aq[4], ak[4];
  {
    const float* rq = Wq + (wave * 16 + l15) * C_;
    const float* rk = Wk + (wave * 16 + l15) * C_;
#pragma unroll
    for (int kt = 0; kt < 4; ++kt) {
      aq[kt] = afrag_global(rq, kt, lhalf);
      ak[kt] = afrag_global(rk, kt, lhalf);
    }
  }
  // per-lane BN affine + graph-context coefficients for its 8 D-rows
  float sQ[8], tQ[8], sK[8], tK[8], g0[8], g1[8], g2[8], g3[8], gb[8];
#pragma unroll
  for (int r = 0; r < 8; ++r) {
    const int o = wave * 16 + r + (lhalf ? 8 : 0);
    const float s0 = gq[o] * rsqrtf(vq[o] + EPS_);
    sQ[r] = s0; tQ[r] = bq[o] - mq[o] * s0;
    const float s1 = gk[o] * rsqrtf(vk[o] + EPS_);
    sK[r] = s1; tK[r] = bk[o] - mk[o] * s1;
    g0[r] = Wg1[o * 2 + 0]; g1[r] = Wg1[o * 2 + 1];
    g2[r] = Wg2[o * 2 + 0]; g3[r] = Wg2[o * 2 + 1];
    gb[r] = bg1[o] + bg2[o];
  }
  const float invtemp = 0.08838834764831845f;   // 1/sqrt(128)

  v8f acc[8];
#pragma unroll
  for (int dt = 0; dt < 8; ++dt) acc[dt] = (v8f){};

  float ld[16];
  load_pc_regs(psrc, ld);                        // chunk 0 in flight

  for (int it = 0; it < CPW; ++it) {
    const int nb = (blockIdx.x * CPW + it) * CHUNK;
    _Float16* buf = pcht[it & 1];

    store_pc_lds(&buf[lane * PCH_STRIDE + wave * 16], ld);
    if (it + 1 < CPW)
      load_pc_regs(psrc + (it + 1) * CHUNK, ld); // next chunk flies over both GEMM stages
    __syncthreads();

    // stage 1: q / (k+gc) for o-tile = wave, two 16-col subtiles
#pragma unroll
    for (int s = 0; s < 2; ++s) {
      v16h bfr[4];
#pragma unroll
      for (int kt = 0; kt < 4; ++kt)
        bfr[kt] = bfrag_lds(&buf[(s * 16 + l15) * PCH_STRIDE + kt * 32 + lhalf * 16]);
      v8f qa = (v8f){}, ka = (v8f){};
#pragma unroll
      for (int kt = 0; kt < 4; ++kt) {
        qa = wmma16(aq[kt], bfr[kt], qa);
        ka = wmma16(ak[kt], bfr[kt], ka);
      }
      const int n  = nb + s * 16 + l15;
      const float x0 = xb[0 * N_ + n], x1 = xb[1 * N_ + n];
      const float x2 = xb[2 * N_ + n], x3 = xb[3 * N_ + n];
#pragma unroll
      for (int r = 0; r < 8; ++r) {
        float qv = fmaxf(qa[r] * sQ[r] + tQ[r], 0.0f) * invtemp;
        float kv = fmaxf(ka[r] * sK[r] + tK[r], 0.0f);
        kv += g0[r] * x0 + g1[r] * x1 + g2[r] * x2 + g3[r] * x3 + gb[r];
        const int row = wave * 16 + r + (lhalf ? 8 : 0);
        const int col = s * 16 + l15;
        qlds [row * QK_STRIDE + col] = (_Float16)qv;
        kglds[row * QK_STRIDE + col] = (_Float16)kv;
      }
    }
    __syncthreads();

    // stage 2: attn tile row ct = wave, contract K = 32 (this chunk)
    {
      v16h af = afrag_lds(&qlds[(wave * 16 + l15) * QK_STRIDE], lhalf);
#pragma unroll
      for (int g = 0; g < 2; ++g) {
        v16h bgrp[4];
#pragma unroll
        for (int j = 0; j < 4; ++j)
          bgrp[j] = bfrag_lds(&kglds[((g * 4 + j) * 16 + l15) * QK_STRIDE + lhalf * 16]);
#pragma unroll
        for (int j = 0; j < 4; ++j)
          acc[g * 4 + j] = wmma16(af, bgrp[j], acc[g * 4 + j]);
      }
    }
    // no barrier here: next iteration stores into the other pcht buffer, and
    // its __syncthreads() separates these kglds reads from the next writes
  }

  float* ab = attn + b * C_ * C_;
#pragma unroll
  for (int dt = 0; dt < 8; ++dt) {
#pragma unroll
    for (int r = 0; r < 8; ++r) {
      const int row = wave * 16 + r + (lhalf ? 8 : 0);
      const int col = dt * 16 + l15;
      atomicAdd(&ab[row * C_ + col], acc[dt][r]);
    }
  }
}

// ---------------------------------------------------------------------------
// Kernel 2: row softmax over d (128) for 1024 rows, in place
// ---------------------------------------------------------------------------
__global__ __launch_bounds__(128)
void CPT_softmax(float* __restrict__ attn)
{
  float* p = attn + (size_t)blockIdx.x * C_;
  const int t = threadIdx.x;
  __shared__ float red[C_];
  const float v = p[t];
  red[t] = v; __syncthreads();
  for (int off = 64; off > 0; off >>= 1) {
    if (t < off) red[t] = fmaxf(red[t], red[t + off]);
    __syncthreads();
  }
  const float mx = red[0]; __syncthreads();
  const float e = __expf(v - mx);
  red[t] = e; __syncthreads();
  for (int off = 64; off > 0; off >>= 1) {
    if (t < off) red[t] += red[t + off];
    __syncthreads();
  }
  p[t] = e / red[0];
}

// ---------------------------------------------------------------------------
// Kernel 3: fused v projection + BN + ReLU + out = attn @ v
// ---------------------------------------------------------------------------
__global__ __launch_bounds__(256)
void CPT_out(const float* __restrict__ pc,
             const float* __restrict__ Wv, const float* __restrict__ gv,
             const float* __restrict__ bv, const float* __restrict__ mv,
             const float* __restrict__ vv,
             const float* __restrict__ attn, float* __restrict__ out)
{
  __shared__ _Float16 pcht [2][CHUNK * PCH_STRIDE]; // pc chunk, K(=c)-major [n][c]
  __shared__ _Float16 vldst[CHUNK * PCH_STRIDE];    // v chunk,  K(=d)-major [n][d]

  const int b     = blockIdx.y;
  const int tid   = threadIdx.x;
  const int wave  = tid >> 5;
  const int lane  = tid & 31;
  const int lhalf = lane >> 4;
  const int l15   = lane & 15;

  const float* pcb = pc  + (size_t)b * C_ * N_;
  float*       ob  = out + (size_t)b * C_ * N_;
  const float* psrc = pcb + (size_t)(wave * 16) * N_ + (size_t)blockIdx.x * CPW * CHUNK + lane;

  // Wv A-fragments for d rows [16*wave, 16*wave+16)
  v16h av[4];
  {
    const float* rv = Wv + (wave * 16 + l15) * C_;
#pragma unroll
    for (int kt = 0; kt < 4; ++kt) av[kt] = afrag_global(rv, kt, lhalf);
  }
  float sV[8], tV[8];
#pragma unroll
  for (int r = 0; r < 8; ++r) {
    const int o = wave * 16 + r + (lhalf ? 8 : 0);
    const float s = gv[o] * rsqrtf(vv[o] + EPS_);
    sV[r] = s; tV[r] = bv[o] - mv[o] * s;
  }
  // softmaxed-attn A-fragments for c rows [16*wave, 16*wave+16), K = d
  v16h aa[4];
  {
    const float* ra = attn + b * C_ * C_ + (wave * 16 + l15) * C_;
#pragma unroll
    for (int kt = 0; kt < 4; ++kt) aa[kt] = afrag_global(ra, kt, lhalf);
  }

  float ld[16];
  load_pc_regs(psrc, ld);

  for (int it = 0; it < CPW; ++it) {
    const int nb = (blockIdx.x * CPW + it) * CHUNK;
    _Float16* buf = pcht[it & 1];

    store_pc_lds(&buf[lane * PCH_STRIDE + wave * 16], ld);
    if (it + 1 < CPW)
      load_pc_regs(psrc + (it + 1) * CHUNK, ld);
    __syncthreads();

    // v rows for d-tile = wave; packed b128 store into K-major vldst
#pragma unroll
    for (int s = 0; s < 2; ++s) {
      v16h bfr[4];
#pragma unroll
      for (int kt = 0; kt < 4; ++kt)
        bfr[kt] = bfrag_lds(&buf[(s * 16 + l15) * PCH_STRIDE + kt * 32 + lhalf * 16]);
      v8f va = (v8f){};
#pragma unroll
      for (int kt = 0; kt < 4; ++kt) va = wmma16(av[kt], bfr[kt], va);
      v8h pk;
#pragma unroll
      for (int r = 0; r < 8; ++r)
        pk[r] = (_Float16)fmaxf(va[r] * sV[r] + tV[r], 0.0f);
      // element r has d = 16*wave + 8*lhalf + r -> one contiguous b128 run
      *(v8h*)&vldst[(s * 16 + l15) * PCH_STRIDE + wave * 16 + lhalf * 8] = pk;
    }
    __syncthreads();

    // out tile row ct = wave : contract K = d (128) against vldst
#pragma unroll
    for (int s = 0; s < 2; ++s) {
      v16h bfr[4];
#pragma unroll
      for (int kt = 0; kt < 4; ++kt)
        bfr[kt] = bfrag_lds(&vldst[(s * 16 + l15) * PCH_STRIDE + kt * 32 + lhalf * 16]);
      v8f oa = (v8f){};
#pragma unroll
      for (int kt = 0; kt < 4; ++kt) oa = wmma16(aa[kt], bfr[kt], oa);
#pragma unroll
      for (int r = 0; r < 8; ++r) {
        const int c = wave * 16 + r + (lhalf ? 8 : 0);
        ob[c * N_ + nb + s * 16 + l15] = oa[r];
      }
    }
    // no barrier: next iteration's __syncthreads() (after ping-pong store)
    // separates these vldst reads from the next v-stage writes
  }
}

// ---------------------------------------------------------------------------
extern "C" void kernel_launch(void* const* d_in, const int* in_sizes, int n_in,
                              void* d_out, int out_size, void* d_ws, size_t ws_size,
                              hipStream_t stream)
{
  (void)in_sizes; (void)n_in; (void)out_size; (void)ws_size;
  const float* pc  = (const float*)d_in[0];
  const float* x   = (const float*)d_in[1];
  const float* Wq  = (const float*)d_in[2];
  const float* gq  = (const float*)d_in[3];
  const float* bq  = (const float*)d_in[4];
  const float* mq  = (const float*)d_in[5];
  const float* vq  = (const float*)d_in[6];
  const float* Wk  = (const float*)d_in[7];
  const float* gk  = (const float*)d_in[8];
  const float* bk  = (const float*)d_in[9];
  const float* mk  = (const float*)d_in[10];
  const float* vk  = (const float*)d_in[11];
  const float* Wv  = (const float*)d_in[12];
  const float* gv  = (const float*)d_in[13];
  const float* bv  = (const float*)d_in[14];
  const float* mv  = (const float*)d_in[15];
  const float* vvp = (const float*)d_in[16];
  const float* Wg1 = (const float*)d_in[17];
  const float* bg1 = (const float*)d_in[18];
  const float* Wg2 = (const float*)d_in[19];
  const float* bg2 = (const float*)d_in[20];

  float* attn = (float*)d_ws;                       // 8*128*128 f32 = 512 KB
  hipMemsetAsync(attn, 0, (size_t)B_ * C_ * C_ * sizeof(float), stream);

  CPT_attn_accum<<<dim3(125, B_), 256, 0, stream>>>(
      pc, x, Wq, gq, bq, mq, vq, Wk, gk, bk, mk, vk, Wg1, bg1, Wg2, bg2, attn);

  CPT_softmax<<<dim3(B_ * C_), 128, 0, stream>>>(attn);

  CPT_out<<<dim3(125, B_), 256, 0, stream>>>(
      pc, Wv, gv, bv, mv, vvp, attn, (float*)d_out);
}
